// ManualGRU_53996328845817
// MI455X (gfx1250) — matrix operand order repaired and checked by reading, not compiled, not run on hardware
//
#include <hip/hip_runtime.h>

// Fused GRU for MI455X (gfx1250, wave32, WMMA).
// Grid: 16 blocks (one per 16-row batch tile), 768 threads = 24 waves.
// Each wave owns one 16-column tile of the 384 gate outputs and keeps its
// W_in / W_h bf16 WMMA fragments resident in VGPRs for all 2048 steps.

typedef __attribute__((ext_vector_type(16))) __bf16 v16bf;
typedef __attribute__((ext_vector_type(8)))  __bf16 v8bf;
typedef __attribute__((ext_vector_type(8)))  float  v8f;

#define Bsz 256
#define Ssz 2048
#define Isz 64
#define Hsz 128
#define N3  384
#define MB  16            // batch rows per workgroup (WMMA M)
#define NTHREADS 768      // 24 waves = 24 N-tiles of 16 columns
#define HST 136           // hbuf row stride in bf16 (272B, conflict-free + 16B aligned)
#define FST 392           // fh row stride in floats
#define XST 136           // xnb row stride in floats

__global__ __launch_bounds__(NTHREADS, 1)
void gru_fused_kernel(const float* __restrict__ inputs,
                      const float* __restrict__ W_in,
                      const float* __restrict__ W_h,
                      const float* __restrict__ bias,
                      float* __restrict__ out)
{
    __shared__ __align__(16) __bf16 hbuf[MB * HST];  // current h, bf16 (A for WMMA)
    __shared__ __align__(16) float  fh[MB * FST];    // gate pre-activations (r,z summed; hn)
    __shared__ __align__(16) float  xnb[MB * XST];   // xn kept separate for the n-gate

    const int tid   = threadIdx.x;
    const int lane  = tid & 31;
    const int hi    = lane >> 4;         // half-wave select
    const int rr    = lane & 15;
    const int b0    = blockIdx.x * MB;   // first batch row of this tile

    // Wave id as a *scalar* so tile-kind branches compile to s_cbranch
    // (no exec-mask save/restore in the hot loop).
    const int wu    = __builtin_amdgcn_readfirstlane(tid) >> 5;   // 0..23, SGPR
    const int col16 = wu * 16 + rr;      // output column in [0,384)

    // ---- loop-invariant weight fragments (held in VGPRs) --------------------
    // B-matrix layout (16-bit, 32x16): element j of v16bf <-> K = 32c + 8*hi + j + (j&8),
    // column N = lane&15.
    v16bf bin0, bin1;                    // W_in: K = 0..31, 32..63
#pragma unroll
    for (int j = 0; j < 16; ++j) {
        bin0[j] = (__bf16)W_in[(8 * hi + j + (j & 8))      * N3 + col16];
        bin1[j] = (__bf16)W_in[(32 + 8 * hi + j + (j & 8)) * N3 + col16];
    }
    v16bf bh[4];                         // W_h: K = 0..127 in 4 chunks of 32
#pragma unroll
    for (int c = 0; c < 4; ++c)
#pragma unroll
        for (int j = 0; j < 16; ++j)
            bh[c][j] = (__bf16)W_h[(32 * c + 8 * hi + j + (j & 8)) * N3 + col16];

    // Biases as loop-invariant splat vectors: used directly as the C operand
    // of the first WMMA in each accumulation chain (zero-cost bias add).
    const float cinx = bias[col16];          // b_in[col]
    const float cinh = bias[N3 + col16];     // b_h[col]
    v8f cinxv, cinhv;
#pragma unroll
    for (int v = 0; v < 8; ++v) { cinxv[v] = cinx; cinhv[v] = cinh; }

    // Per-lane pointer to row rr of this batch tile in `inputs` (advances 64 f32/step).
    const float* px = inputs + ((size_t)(b0 + rr) * Ssz) * Isz;

    // ---- elementwise ownership: e = tid + 768*k (k=0,1 full, k=2 tid<512) ---
    float hstate[3];
    int   eoffF[3], eoffX[3], eoffH[3];
    float* outp[3];
#pragma unroll
    for (int k = 0; k < 3; ++k) {
        int e   = tid + NTHREADS * k;    // < 2048 for k<2 always; k=2 iff tid<512
        int r_  = (e >> 7) & 15;
        int c_  = e & 127;
        eoffF[k] = r_ * FST + c_;
        eoffX[k] = r_ * XST + c_;
        eoffH[k] = r_ * HST + c_;
        hstate[k] = 0.0f;
        outp[k] = out + ((size_t)(b0 + r_) * Ssz) * Hsz + c_;
    }
    const bool has3 = (tid < MB * Hsz - 2 * NTHREADS);   // tid < 512

    for (int i = tid; i < MB * HST; i += NTHREADS) hbuf[i] = (__bf16)0.0f;
    __syncthreads();

    const bool is_n_tile = (wu >= 16);               // scalar -> uniform branch
    const int  coln      = (wu - 16) * 16 + rr;      // column within xn (n-tiles only)
    const int  base0     = 8 * hi;                   // A fragment K offsets
    const int  rbase     = 8 * hi;                   // C fragment row base

    for (int s = 0; s < Ssz; ++s) {
        // ---- A_x fragments: global fp32 -> bf16 (rows shared via WGP$) ------
        // A layout (16-bit, 16x32): elem j <-> K = 32c+8hi+j, elem 8+j <-> +16.
        v16bf ax0, ax1;
#pragma unroll
        for (int j = 0; j < 8; ++j) {
            ax0[j]     = (__bf16)px[base0 + j];
            ax0[8 + j] = (__bf16)px[base0 + 16 + j];
            ax1[j]     = (__bf16)px[32 + base0 + j];
            ax1[8 + j] = (__bf16)px[32 + base0 + 16 + j];
        }

        // x-projection: 2 WMMAs (K = 64), bias vector as initial C
        v8f cx = __builtin_amdgcn_wmma_f32_16x16x32_bf16(false, ax0, false, bin0,
                                                         (short)0, cinxv, false, false);
        cx = __builtin_amdgcn_wmma_f32_16x16x32_bf16(false, ax1, false, bin1,
                                                     (short)0, cx, false, false);

        // h @ W_h: 4 WMMAs (K = 128), A from LDS (two ds_load_b128 per chunk)
        v8f ch = cinhv;
#pragma unroll
        for (int c = 0; c < 4; ++c) {
            const __bf16* p = &hbuf[rr * HST + 32 * c + base0];
            v8bf lo  = *(const v8bf*)p;
            v8bf hi8 = *(const v8bf*)(p + 16);
            v16bf ah;
#pragma unroll
            for (int j = 0; j < 8; ++j) { ah[j] = lo[j]; ah[8 + j] = hi8[j]; }
            ch = __builtin_amdgcn_wmma_f32_16x16x32_bf16(false, ah, false, bh[c],
                                                         (short)0, ch, false, false);
        }

        // ---- scatter gate pre-activations to LDS (uniform branch) -----------
        // C/D layout: elem v <-> row = v + 8*hi, col = lane&15.
        if (!is_n_tile) {
#pragma unroll
            for (int v = 0; v < 8; ++v)
                fh[(rbase + v) * FST + col16] = cx[v] + ch[v];  // xr+hr(+b) / xz+hz(+b)
        } else {
#pragma unroll
            for (int v = 0; v < 8; ++v) {
                fh[(rbase + v) * FST + col16] = ch[v];          // hn + b_h
                xnb[(rbase + v) * XST + coln] = cx[v];          // xn + b_in
            }
        }
        __syncthreads();

        // ---- gates + state update ------------------------------------------
#pragma unroll
        for (int k = 0; k < 3; ++k) {
            if (k < 2 || has3) {
                float sr  = fh[eoffF[k]];
                float sz  = fh[eoffF[k] + 128];
                float hn  = fh[eoffF[k] + 256];
                float xnv = xnb[eoffX[k]];
                float rg = 1.0f / (1.0f + __expf(-sr));
                float zg = 1.0f / (1.0f + __expf(-sz));
                float xt = xnv + rg * hn;
                xt = fminf(fmaxf(xt, -20.0f), 20.0f);
                float t  = __expf(-2.0f * xt);
                float ng = (1.0f - t) / (1.0f + t);      // tanh
                float hN = (1.0f - zg) * ng + zg * hstate[k];
                hstate[k] = hN;
                *outp[k] = hN;                           // hidden_seq[b, s, col]
                outp[k] += Hsz;
                hbuf[eoffH[k]] = (__bf16)hN;             // next step's A matrix
            }
        }
        __syncthreads();
        px += Isz;
    }

    // ---- h_last -------------------------------------------------------------
#pragma unroll
    for (int k = 0; k < 3; ++k) {
        if (k < 2 || has3) {
            int e  = tid + NTHREADS * k;
            int r_ = (e >> 7) & 15;
            int c_ = e & 127;
            out[(size_t)Bsz * Ssz * Hsz + (size_t)(b0 + r_) * Hsz + c_] = hstate[k];
        }
    }
}

extern "C" void kernel_launch(void* const* d_in, const int* in_sizes, int n_in,
                              void* d_out, int out_size, void* d_ws, size_t ws_size,
                              hipStream_t stream)
{
    const float* inputs = (const float*)d_in[0];   // (256, 2048, 64)
    const float* W_in   = (const float*)d_in[1];   // (64, 384)
    const float* W_h    = (const float*)d_in[2];   // (128, 384)
    const float* bias   = (const float*)d_in[3];   // (768,)
    float* out = (float*)d_out;                    // hidden_seq (256,2048,128) ++ h_last (256,128)

    gru_fused_kernel<<<Bsz / MB, NTHREADS, 0, stream>>>(inputs, W_in, W_h, bias, out);
}
